// MoELayer_41996190220294
// MI455X (gfx1250) — compile-verified, matrix operand order checked by reading, not run on hardware
//
#include <hip/hip_runtime.h>

typedef _Float16 v16h __attribute__((ext_vector_type(16)));
typedef _Float16 v8h  __attribute__((ext_vector_type(8)));
typedef _Float16 v4h  __attribute__((ext_vector_type(4)));
typedef float    v8f  __attribute__((ext_vector_type(8)));
typedef float    v4f  __attribute__((ext_vector_type(4)));

constexpr int T = 512;      // tokens (B*S)
constexpr int H = 2048;     // hidden
constexpr int I = 1536;     // moe intermediate
constexpr int E = 32;       // routed experts
constexpr int K = 4;        // experts per token
constexpr int SLOTS = T * K;
constexpr float SCALE = 1.8f;

// ---------------------------------------------------------------- router ----
__global__ void router_kernel(const float* __restrict__ x, const float* __restrict__ rw,
                              const float* __restrict__ rb, int* __restrict__ topk_idx,
                              float* __restrict__ topk_w) {
  __shared__ float xs[H];
  __shared__ float lg[E];
  const int t = blockIdx.x;
  const int tid = threadIdx.x;
  for (int i = tid; i < H; i += 256) xs[i] = x[(size_t)t * H + i];
  __syncthreads();
  const int e = tid >> 3, l = tid & 7;           // 8 lanes per expert
  const float* wrow = rw + (size_t)e * H;
  float s = 0.f;
  for (int k = l; k < H; k += 8) s += xs[k] * wrow[k];
  s += __shfl_xor(s, 1); s += __shfl_xor(s, 2); s += __shfl_xor(s, 4);
  if (l == 0) lg[e] = s + rb[e];
  __syncthreads();
  if (tid == 0) {
    float v[E];
    for (int i = 0; i < E; ++i) v[i] = lg[i];
    int bi[K]; float bv[K];
    for (int j = 0; j < K; ++j) {
      int best = 0; float bm = v[0];
      for (int i = 1; i < E; ++i) if (v[i] > bm) { bm = v[i]; best = i; }
      bi[j] = best; bv[j] = bm; v[best] = -3.4e38f;
    }
    const float mx = bv[0];                       // first pick is the max
    float ssum = 0.f, ex[K];
    for (int j = 0; j < K; ++j) { ex[j] = __expf(bv[j] - mx); ssum += ex[j]; }
    for (int j = 0; j < K; ++j) {
      topk_idx[t * K + j] = bi[j];
      topk_w[t * K + j]   = SCALE * ex[j] / ssum;
    }
  }
}

// ---------------------------------------------------------------- bucket ----
__global__ void bucket_kernel(const int* __restrict__ topk_idx, const float* __restrict__ topk_w,
                              int* __restrict__ cnt, int* __restrict__ offs,
                              int* __restrict__ slot_token, float* __restrict__ slot_w,
                              int* __restrict__ slot_of, float* __restrict__ ones) {
  __shared__ int scnt[E];
  __shared__ int soff[E];
  const int tid = threadIdx.x;
  for (int i = tid; i < T + 16; i += 256) ones[i] = 1.0f;   // weight vec for shared expert
  if (tid < E) scnt[tid] = 0;
  __syncthreads();
  for (int i = tid; i < SLOTS; i += 256) atomicAdd(&scnt[topk_idx[i]], 1);
  __syncthreads();
  if (tid == 0) {
    int run = 0;
    for (int e = 0; e < E; ++e) { soff[e] = run; offs[e] = run; cnt[e] = scnt[e]; run += scnt[e]; }
  }
  __syncthreads();
  if (tid < E) {                                   // deterministic scatter, token order
    int p = soff[tid];
    for (int i = 0; i < SLOTS; ++i) {
      if (topk_idx[i] == tid) {
        slot_token[p] = i >> 2;
        slot_w[p]     = topk_w[i];
        slot_of[i]    = p;
        ++p;
      }
    }
  }
}

// -------------------------------------------------------- gate/up (WMMA) ----
// Block: 256 threads = 8 waves; each wave owns a 16x16 tile of (gate,up).
// grid.x = I/128, grid.y = token tiles, grid.z = expert (or 1 in shared mode).
// wvec = per-slot combine weight (routed) or an all-ones vector (shared) ->
// branchless epilogue.
__global__ void __launch_bounds__(256)
gateup_kernel(const float* __restrict__ x, const float* __restrict__ gw,
              const float* __restrict__ uw, const int* __restrict__ cnt,
              const int* __restrict__ offs, const int* __restrict__ slot_token,
              const float* __restrict__ wvec, _Float16* __restrict__ act_out,
              int shared_mode) {
  __shared__ __align__(128) _Float16 xs[16 * H];   // 64 KB X tile (f16)
  const int e = blockIdx.z;
  const int mt = blockIdx.y;
  const int nchunk = blockIdx.x;
  const int count = shared_mode ? T : cnt[e];
  if (mt * 16 >= count) return;
  const int off_e = shared_mode ? 0 : offs[e];
  const float* gwe = shared_mode ? gw : gw + (size_t)e * I * H;
  const float* uwe = shared_mode ? uw : uw + (size_t)e * I * H;
  const int tid = threadIdx.x;

  {   // stage 16 tokens of x into LDS as f16 (pad rows -> zeros), float4 vectorized
    const int row  = tid >> 4;
    const int srow = mt * 16 + row;
    const bool valid = srow < count;
    int tok = 0;
    if (valid) tok = shared_mode ? srow : slot_token[off_e + srow];
    const float* xr = x + (size_t)tok * H;
    for (int c = (tid & 15) * 4; c < H; c += 64) {
      v4f xv = valid ? *(const v4f*)(xr + c) : (v4f){0.f, 0.f, 0.f, 0.f};
      v4h hv;
#pragma unroll
      for (int j = 0; j < 4; ++j) hv[j] = (_Float16)xv[j];
      *(v4h*)(xs + row * H + c) = hv;
    }
  }
  __syncthreads();

  const int wave = tid >> 5, lane = tid & 31;
  const int half = lane >> 4, nl = lane & 15;
  const int col = nchunk * 128 + wave * 16 + nl;   // column of I
  const float* grow = gwe + (size_t)col * H;
  const float* urow = uwe + (size_t)col * H;
  const _Float16* xrow = xs + (size_t)nl * H;

  v8f cg = {}; v8f cu = {};
#pragma unroll 2
  for (int kb = 0; kb < H; kb += 32) {
    const int k0 = kb + half * 8, k1 = kb + 16 + half * 8;
    __builtin_prefetch(grow + kb + 256, 0, 1);     // speculative: OOB is dropped
    __builtin_prefetch(urow + kb + 256, 0, 1);
    v8h a0 = *(const v8h*)(xrow + k0);
    v8h a1 = *(const v8h*)(xrow + k1);
    v16h a, bg, bu;
#pragma unroll
    for (int j = 0; j < 8; ++j) { a[j] = a0[j]; a[8 + j] = a1[j]; }
#pragma unroll
    for (int j = 0; j < 8; ++j) {
      bg[j]     = (_Float16)grow[k0 + j];
      bg[8 + j] = (_Float16)grow[k1 + j];
      bu[j]     = (_Float16)urow[k0 + j];
      bu[8 + j] = (_Float16)urow[k1 + j];
    }
    cg = __builtin_amdgcn_wmma_f32_16x16x32_f16(false, a, false, bg, (short)0, cg, false, false);
    cu = __builtin_amdgcn_wmma_f32_16x16x32_f16(false, a, false, bu, (short)0, cu, false, false);
  }

  // Branchless epilogue: silu(g)*u * weight for all 8 elements; only the final
  // store is predicated (exec-masked). Pad rows read in-workspace weight bytes.
  const float* wrow = wvec + off_e + mt * 16 + half * 8;  // 8 consecutive weights
  float wv[8];
#pragma unroll
  for (int r = 0; r < 8; ++r) wv[r] = wrow[r];
  _Float16 hv[8];
#pragma unroll
  for (int r = 0; r < 8; ++r) {                    // C layout: m = 8*half + r, n = lane&15
    const float g = cg[r], u = cu[r];
    const float h = g * __builtin_amdgcn_rcpf(1.f + __expf(-g)) * u * wv[r];
    hv[r] = (_Float16)h;
  }
#pragma unroll
  for (int r = 0; r < 8; ++r) {
    const int srow = mt * 16 + half * 8 + r;
    if (srow < count)
      act_out[(size_t)(off_e + srow) * I + col] = hv[r];  // off_e==0 in shared mode
  }
}

// -------------------------------------------------------- down-proj (WMMA) --
// grid.x = H/128, grid.y = token tiles, grid.z = expert (or 1 in shared mode).
// Activation tile is raw f16 -> staged with GLOBAL_LOAD_ASYNC_TO_LDS_B128.
__global__ void __launch_bounds__(256)
down_kernel(const _Float16* __restrict__ act, const float* __restrict__ dw,
            const int* __restrict__ cnt, const int* __restrict__ offs,
            float* __restrict__ outp, int shared_mode) {
  constexpr int LP = I + 8;                        // padded LDS row stride (3088 B, 16B-mult)
  __shared__ __align__(128) _Float16 as[16 * LP];  // ~48 KB
  const int e = blockIdx.z, mt = blockIdx.y, nchunk = blockIdx.x;
  const int count = shared_mode ? T : cnt[e];
  if (mt * 16 >= count) return;
  const int off_e = shared_mode ? 0 : offs[e];
  const float* dwe = shared_mode ? dw : dw + (size_t)e * H * I;
  const int tid = threadIdx.x;

  {   // async-stage activation tile into LDS: 16 rows x 3072 B, 16 B per lane per issue.
      // Rows past `count` copy deterministic neighboring act bytes; their WMMA output
      // rows are never stored, so garbage there is harmless.
    const int row = tid >> 4;                      // 16 threads per row
    const int srow = mt * 16 + row;
    const char* grow_b = (const char*)(act + (size_t)(off_e + srow) * I);
    const unsigned lrow = (unsigned)(uintptr_t)(as + (size_t)row * LP);
    for (int cb = (tid & 15) * 16; cb < I * 2; cb += 256) {
      unsigned lofs = lrow + (unsigned)cb;
      unsigned long long gaddr = (unsigned long long)(uintptr_t)(grow_b + cb);
      asm volatile("global_load_async_to_lds_b128 %0, %1, off"
                   :: "v"(lofs), "v"(gaddr) : "memory");
    }
    asm volatile("s_wait_asynccnt 0" ::: "memory");
  }
  __syncthreads();

  const int wave = tid >> 5, lane = tid & 31;
  const int half = lane >> 4, nl = lane & 15;
  const int col = nchunk * 128 + wave * 16 + nl;   // column of H
  const float* drow = dwe + (size_t)col * I;
  const _Float16* arow = as + (size_t)nl * LP;

  v8f c = {};
#pragma unroll 2
  for (int kb = 0; kb < I; kb += 32) {
    const int k0 = kb + half * 8, k1 = kb + 16 + half * 8;
    __builtin_prefetch(drow + kb + 256, 0, 1);     // speculative: OOB is dropped
    v8h a0 = *(const v8h*)(arow + k0);
    v8h a1 = *(const v8h*)(arow + k1);
    v16h a, b;
#pragma unroll
    for (int j = 0; j < 8; ++j) { a[j] = a0[j]; a[8 + j] = a1[j]; }
#pragma unroll
    for (int j = 0; j < 8; ++j) {
      b[j]     = (_Float16)drow[k0 + j];
      b[8 + j] = (_Float16)drow[k1 + j];
    }
    c = __builtin_amdgcn_wmma_f32_16x16x32_f16(false, a, false, b, (short)0, c, false, false);
  }

#pragma unroll
  for (int r = 0; r < 8; ++r) {                    // store-only predication
    const int srow = mt * 16 + half * 8 + r;
    if (srow < count)
      outp[(size_t)(off_e + srow) * H + col] = c[r];  // off_e==0 in shared mode -> d_out
  }
}

// --------------------------------------------------------------- combine ----
__global__ void combine_kernel(const float* __restrict__ routed, const int* __restrict__ slot_of,
                               float* __restrict__ out) {
  const int t = blockIdx.x;
  const int s0 = slot_of[t * K + 0], s1 = slot_of[t * K + 1];
  const int s2 = slot_of[t * K + 2], s3 = slot_of[t * K + 3];
  const float* r0 = routed + (size_t)s0 * H;
  const float* r1 = routed + (size_t)s1 * H;
  const float* r2 = routed + (size_t)s2 * H;
  const float* r3 = routed + (size_t)s3 * H;
  for (int h = threadIdx.x; h < H; h += 256)
    out[(size_t)t * H + h] += r0[h] + r1[h] + r2[h] + r3[h];
}

// ---------------------------------------------------------------- launch ----
extern "C" void kernel_launch(void* const* d_in, const int* in_sizes, int n_in,
                              void* d_out, int out_size, void* d_ws, size_t ws_size,
                              hipStream_t stream) {
  const float* x   = (const float*)d_in[0];
  const float* rw  = (const float*)d_in[1];
  const float* rb  = (const float*)d_in[2];
  const float* gw  = (const float*)d_in[3];
  const float* uw  = (const float*)d_in[4];
  const float* dwn = (const float*)d_in[5];
  const float* sgw = (const float*)d_in[6];
  const float* suw = (const float*)d_in[7];
  const float* sdw = (const float*)d_in[8];
  float* out = (float*)d_out;

  char* ws = (char*)d_ws;
  int*      topk_idx   = (int*)     (ws + 0);          //  8 KB
  float*    topk_w     = (float*)   (ws + 8192);       //  8 KB
  int*      cnt        = (int*)     (ws + 16384);      //  128 B
  int*      offs       = (int*)     (ws + 16640);      //  128 B
  int*      slot_token = (int*)     (ws + 16896);      //  8 KB
  float*    slot_w     = (float*)   (ws + 25088);      //  8 KB
  int*      slot_of    = (int*)     (ws + 33280);      //  8 KB
  float*    ones       = (float*)   (ws + 41472);      //  ~2.1 KB (T+16 ones)
  _Float16* act        = (_Float16*)(ws + 43776);      //  6.29 MB (SLOTS x I, f16)
  _Float16* sact       = (_Float16*)(ws + 6335232);    //  1.57 MB (T x I, f16)
  float*    routed     = (float*)   (ws + 7908096);    // 16.78 MB (SLOTS x H, f32)
  (void)in_sizes; (void)n_in; (void)out_size; (void)ws_size;

  router_kernel <<<T, 256, 0, stream>>>(x, rw, rb, topk_idx, topk_w);
  bucket_kernel <<<1, 256, 0, stream>>>(topk_idx, topk_w, cnt, offs, slot_token, slot_w, slot_of, ones);
  gateup_kernel <<<dim3(I / 128, T / 16, E), 256, 0, stream>>>(x, gw,  uw,  cnt, offs, slot_token, slot_w, act,  0);
  gateup_kernel <<<dim3(I / 128, T / 16, 1), 256, 0, stream>>>(x, sgw, suw, cnt, offs, slot_token, ones,   sact, 1);
  down_kernel   <<<dim3(H / 128, T / 16, 1), 256, 0, stream>>>(sact, sdw, cnt, offs, out,    1);
  down_kernel   <<<dim3(H / 128, T / 16, E), 256, 0, stream>>>(act,  dwn, cnt, offs, routed, 0);
  combine_kernel<<<T, 256, 0, stream>>>(routed, slot_of, out);
}
